// GRUDisentangledAttention_15049565405470
// MI455X (gfx1250) — matrix operand order repaired and checked by reading, not compiled, not run on hardware
//
#include <hip/hip_runtime.h>
#include <math.h>

// ---------------- constants ----------------
#define BB   8
#define SS   512
#define DD   1024
#define HH   8
#define HD   128
#define FFD  4096
#define GH   512

typedef __attribute__((ext_vector_type(16))) _Float16 v16h;
typedef __attribute__((ext_vector_type(8)))  _Float16 v8h;
typedef __attribute__((ext_vector_type(2)))  _Float16 h2v;
typedef __attribute__((ext_vector_type(8)))  float    v8f;
typedef __attribute__((ext_vector_type(4)))  unsigned int u32x4;
typedef __attribute__((ext_vector_type(8)))  int      i32x8;
typedef __attribute__((ext_vector_type(4)))  int      i32x4;

static __device__ __forceinline__ v16h cat8(v8h lo, v8h hi) {
  return __builtin_shufflevector(lo, hi, 0,1,2,3,4,5,6,7,8,9,10,11,12,13,14,15);
}

#define WMMA16(a, b, c) \
  __builtin_amdgcn_wmma_f32_16x16x32_f16(false, (a), false, (b), (short)0, (c), false, false)

// ---------------- Tensor Data Mover: 2D f16 tile global -> LDS ----------------
// D# per CDNA5 ISA ch.8: group0 = {count, lds_addr, global_addr, type=2},
// group1 = {data_size=2B, tensor_dim0=row_len, tensor_dim1=rows, tile 32xRows,
//           tensor_dim0_stride=row_len}. Groups 2/3 zero (2D tensor).
// Toolchain uses the 6-arg builtin: (u32x4, i32x8, i32x4, i32x4, i32x8, cpol).
static __device__ __forceinline__ void tdm_load_2d(unsigned lds_off,
                                                   const _Float16* gptr,
                                                   unsigned tile_k, unsigned tile_rows,
                                                   unsigned row_stride_elems) {
  unsigned long long ga = (unsigned long long)(size_t)gptr;
  u32x4 g0;
  g0[0] = 1u;                                      // count=1 (valid user descriptor)
  g0[1] = lds_off;                                 // lds_addr (bytes)
  g0[2] = (unsigned)(ga & 0xffffffffu);            // global_addr[31:0]
  g0[3] = (unsigned)((ga >> 32) & 0x01ffffffu) | (2u << 30);  // addr[56:32], type=2
  i32x8 g1;
  unsigned td0 = row_stride_elems;                 // tensor_dim0 (elements)
  unsigned td1 = tile_rows;                        // tensor_dim1
  g1[0] = (int)(1u << 16);                         // wg_mask=0, data_size=1 (2 bytes)
  g1[1] = (int)((td0 & 0xffffu) << 16);            // abar_addr=0 | tensor_dim0[15:0]
  g1[2] = (int)((td0 >> 16) | ((td1 & 0xffffu) << 16));   // dim0 hi | dim1 lo
  g1[3] = (int)((td1 >> 16) | (tile_k << 16));            // dim1 hi | tile_dim0
  g1[4] = (int)(tile_rows & 0xffffu);              // tile_dim1 | tile_dim2=0
  g1[5] = (int)row_stride_elems;                   // tensor_dim0_stride[31:0]
  g1[6] = 0;                                       // stride hi | dim1_stride lo
  g1[7] = 0;
  i32x4 z4 = {0, 0, 0, 0};
  i32x8 z8 = {0, 0, 0, 0, 0, 0, 0, 0};
  __builtin_amdgcn_tensor_load_to_lds(g0, g1, z4, z4, z8, 0);
}

// ---------------- fp32 -> f16 convert ----------------
__global__ __launch_bounds__(256) void k_cvt(const float* __restrict__ src,
                                             _Float16* __restrict__ dst, int n) {
  int i = blockIdx.x * 256 + threadIdx.x;
  if (i < n) dst[i] = (_Float16)src[i];
}

// ---------------- block reduce helper (256 threads / 8 waves) ----------------
static __device__ __forceinline__ float block_sum(float v, float* scratch) {
  #pragma unroll
  for (int o = 16; o > 0; o >>= 1) v += __shfl_xor(v, o, 32);
  int lane = threadIdx.x & 31, w = threadIdx.x >> 5;
  if (lane == 0) scratch[w] = v;
  __syncthreads();
  float tot = 0.f;
  #pragma unroll
  for (int i = 0; i < 8; i++) tot += scratch[i];
  __syncthreads();
  return tot;
}

// ---------------- y = LN(a + b) (+f16 copy) ----------------
__global__ __launch_bounds__(256) void k_add_ln(const float* __restrict__ a,
                                                const float* __restrict__ b,
                                                const float* __restrict__ g,
                                                const float* __restrict__ beta,
                                                float* __restrict__ outf,
                                                _Float16* __restrict__ outh,
                                                int b2_pos) {
  __shared__ float scratch[8];
  int row = blockIdx.x;
  const float* ar = a + (size_t)row * DD;
  const float* br = b + (size_t)(b2_pos ? (row & (SS - 1)) : row) * DD;
  int t = threadIdx.x;
  float v[4]; float s = 0.f;
  #pragma unroll
  for (int j = 0; j < 4; j++) { v[j] = ar[t * 4 + j] + br[t * 4 + j]; s += v[j]; }
  float mean = block_sum(s, scratch) * (1.f / DD);
  float s2 = 0.f;
  #pragma unroll
  for (int j = 0; j < 4; j++) { float d = v[j] - mean; s2 += d * d; }
  float var = block_sum(s2, scratch) * (1.f / DD);
  float rstd = rsqrtf(var + 1e-7f);
  #pragma unroll
  for (int j = 0; j < 4; j++) {
    int c = t * 4 + j;
    float y = (v[j] - mean) * rstd * g[c] + beta[c];
    if (outf) outf[(size_t)row * DD + c] = y;
    if (outh) outh[(size_t)row * DD + c] = (_Float16)y;
  }
}

// ---------------- WMMA GEMM with TDM-staged, double-buffered LDS tiles ----------------
// C[M,N] = act(A[M,K] @ W[N,K]^T + bias); ACT: 0 none, 1 exact gelu
template <int ACT>
__global__ __launch_bounds__(256) void k_gemm(const _Float16* __restrict__ A,
                                              const _Float16* __restrict__ W,
                                              const float* __restrict__ bias,
                                              float* __restrict__ Cf,
                                              _Float16* __restrict__ Ch,
                                              int M, int N, int K) {
  __shared__ __align__(32) _Float16 As[2][64][32];
  __shared__ __align__(32) _Float16 Ws[2][64][32];
  const int bm = blockIdx.x * 64, bn = blockIdx.y * 64;
  const int tid = threadIdx.x;
  const int lane = tid & 31, wave = tid >> 5;
  const int wm = wave & 3, wn = wave >> 2;   // 4 x 2 wave grid: 16 rows x 32 cols each
  v8f acc[2] = {{}, {}};
  const int nk = K / 32;

  const unsigned ldsA0 = (unsigned)(size_t)&As[0][0][0];
  const unsigned ldsA1 = (unsigned)(size_t)&As[1][0][0];
  const unsigned ldsW0 = (unsigned)(size_t)&Ws[0][0][0];
  const unsigned ldsW1 = (unsigned)(size_t)&Ws[1][0][0];

  // prologue: TDM-stage tile 0 (wave 0 drives the DMA engine)
  if (wave == 0) {
    tdm_load_2d(ldsA0, &A[(size_t)bm * K], 32, 64, (unsigned)K);
    tdm_load_2d(ldsW0, &W[(size_t)bn * K], 32, 64, (unsigned)K);
    __builtin_amdgcn_s_wait_tensorcnt(0);
  }
  __syncthreads();

  for (int i = 0; i < nk; i++) {
    const int cur = i & 1;
    // kick DMA for next tile while this tile computes
    if (wave == 0 && (i + 1) < nk) {
      const unsigned la = cur ? ldsA0 : ldsA1;
      const unsigned lw = cur ? ldsW0 : ldsW1;
      tdm_load_2d(la, &A[(size_t)bm * K + (size_t)(i + 1) * 32], 32, 64, (unsigned)K);
      tdm_load_2d(lw, &W[(size_t)bn * K + (size_t)(i + 1) * 32], 32, 64, (unsigned)K);
    }
    // A fragment: row M = wm*16 + lane%16; K-halves split by lane group
    const int mrow = wm * 16 + (lane & 15);
    const int ka = (lane < 16) ? 0 : 16;
    v16h afrag = cat8(*(const v8h*)&As[cur][mrow][ka],
                      *(const v8h*)&As[cur][mrow][ka + 8]);
    // B fragments: col N = lane%16 within subtile; 16 contiguous K per lane
    const int kb = (lane < 16) ? 0 : 16;
    #pragma unroll
    for (int j = 0; j < 2; j++) {
      const int ncol = wn * 32 + j * 16 + (lane & 15);
      v16h bfrag = *(const v16h*)&Ws[cur][ncol][kb];
      acc[j] = WMMA16(afrag, bfrag, acc[j]);
    }
    __syncthreads();                          // everyone done reading buf `cur`
    if (wave == 0) __builtin_amdgcn_s_wait_tensorcnt(0);  // next tile landed
    __syncthreads();
  }

  // epilogue: C/D layout — VGPR r, lane<16 -> M=r, lane>=16 -> M=8+r; N = lane%16
  const int mr = bm + wm * 16 + ((lane >> 4) << 3);
  #pragma unroll
  for (int j = 0; j < 2; j++) {
    const int nc = bn + wn * 32 + j * 16 + (lane & 15);
    const float bval = bias ? bias[nc] : 0.f;
    #pragma unroll
    for (int r = 0; r < 8; r++) {
      float x = acc[j][r] + bval;
      if (ACT == 1) x = 0.5f * x * (1.f + erff(x * 0.70710678118f));
      if (Cf) Cf[(size_t)(mr + r) * N + nc] = x;
      if (Ch) Ch[(size_t)(mr + r) * N + nc] = (_Float16)x;
    }
  }
}

// ---------------- V transpose: vT[b*H+h][d][s] = v16[(b*S+s)*D + h*HD + d] ----------------
__global__ __launch_bounds__(256) void k_vt(const _Float16* __restrict__ v16,
                                            _Float16* __restrict__ vT) {
  int idx = blockIdx.x * 256 + threadIdx.x;        // over B*H*HD*S
  int s = idx & (SS - 1);
  int d = (idx >> 9) & (HD - 1);
  int bh = idx >> 16;
  int b = bh >> 3, h = bh & 7;
  vT[idx] = v16[((size_t)(b * SS + s)) * DD + h * HD + d];
}

// ---------------- fused attention: scores -> softmax -> ctx (per b,h,16-query tile) ----------
__global__ __launch_bounds__(256) void k_attn(const _Float16* __restrict__ q16,
                                              const _Float16* __restrict__ k16,
                                              const _Float16* __restrict__ vT,
                                              _Float16* __restrict__ ctx16) {
  __shared__ __align__(32) _Float16 Qs[16][HD];
  __shared__ float                  Ss[16][SS];
  __shared__ __align__(32) _Float16 Ps[16][SS];
  const int bh = blockIdx.x, qt = blockIdx.y;
  const int b = bh >> 3, h = bh & 7;
  const int tid = threadIdx.x, lane = tid & 31, wave = tid >> 5;

  // stage Q tile (16 x 128 f16)
  {
    int row = tid >> 4, col = (tid & 15) * 8;
    *(uint4*)&Qs[row][col] =
        *(const uint4*)&q16[((size_t)(b * SS + qt * 16 + row)) * DD + h * HD + col];
  }
  __syncthreads();

  // scores: each wave computes 4 key tiles of 16 (8 waves * 4 * 16 = 512 keys)
  const float scale = 0.08838834764831845f;  // 1/sqrt(128)
  for (int kt0 = 0; kt0 < 4; kt0++) {
    const int kt = wave * 4 + kt0;
    v8f acc = {};
    #pragma unroll
    for (int kc = 0; kc < 4; kc++) {                 // HD=128 in K=32 chunks
      const int mrow = lane & 15;
      const int ka = kc * 32 + ((lane < 16) ? 0 : 16);
      v16h afrag = cat8(*(const v8h*)&Qs[mrow][ka], *(const v8h*)&Qs[mrow][ka + 8]);
      const int key = kt * 16 + (lane & 15);
      const int kb = kc * 32 + ((lane < 16) ? 0 : 16);
      v16h bfrag = *(const v16h*)&k16[((size_t)(b * SS + key)) * DD + h * HD + kb];
      acc = WMMA16(afrag, bfrag, acc);
    }
    const int rbase = (lane >> 4) << 3;
    #pragma unroll
    for (int r = 0; r < 8; r++)
      Ss[rbase + r][kt * 16 + (lane & 15)] = acc[r] * scale;
  }
  __syncthreads();

  // softmax: wave w handles rows 2w, 2w+1 with wave32 reductions
  #pragma unroll
  for (int rr = 0; rr < 2; rr++) {
    const int row = wave * 2 + rr;
    float mx = -1e30f;
    for (int j = lane; j < SS; j += 32) mx = fmaxf(mx, Ss[row][j]);
    #pragma unroll
    for (int o = 16; o > 0; o >>= 1) mx = fmaxf(mx, __shfl_xor(mx, o, 32));
    float sum = 0.f;
    for (int j = lane; j < SS; j += 32) {
      float e = __expf(Ss[row][j] - mx);
      Ss[row][j] = e;
      sum += e;
    }
    #pragma unroll
    for (int o = 16; o > 0; o >>= 1) sum += __shfl_xor(sum, o, 32);
    float inv = 1.f / sum;
    for (int j = lane; j < SS; j += 32) Ps[row][j] = (_Float16)(Ss[row][j] * inv);
  }
  __syncthreads();

  // ctx = P[16,512] @ V[512,128]; wave w owns output dims w*16..w*16+15
  v8f acc = {};
  for (int kc = 0; kc < SS / 32; kc++) {
    const int mrow = lane & 15;
    const int ka = kc * 32 + ((lane < 16) ? 0 : 16);
    v16h afrag = cat8(*(const v8h*)&Ps[mrow][ka], *(const v8h*)&Ps[mrow][ka + 8]);
    const int n = wave * 16 + (lane & 15);
    const int kb = kc * 32 + ((lane < 16) ? 0 : 16);
    v16h bfrag = *(const v16h*)&vT[((size_t)bh * HD + n) * SS + kb];
    acc = WMMA16(afrag, bfrag, acc);
  }
  {
    const int rbase = (lane >> 4) << 3;
    const int col = h * HD + wave * 16 + (lane & 15);
    #pragma unroll
    for (int r = 0; r < 8; r++) {
      const int q = qt * 16 + rbase + r;
      ctx16[((size_t)(b * SS + q)) * DD + col] = (_Float16)acc[r];
    }
  }
}

// ---------------- GRU scan: one block per (batch, direction), 512 threads ------------
// Recurrent matvec uses packed-f16 FMAs (v_pk_fma_f16) against L2-resident whh (f16).
__global__ __launch_bounds__(512) void k_gru(const float* __restrict__ gx_f,
                                             const float* __restrict__ gx_b,
                                             const _Float16* __restrict__ whh16_f,
                                             const float* __restrict__ bhh_f,
                                             const _Float16* __restrict__ whh16_b,
                                             const float* __restrict__ bhh_b,
                                             float* __restrict__ out) {
  __shared__ float hs[GH];
  __shared__ __align__(4) _Float16 hs16[GH];
  const int b = blockIdx.x, dir = blockIdx.y;
  const float*    gx  = dir ? gx_b    : gx_f;
  const _Float16* whh = dir ? whh16_b : whh16_f;
  const float*    bhh = dir ? bhh_b   : bhh_f;
  const int i = threadIdx.x;            // hidden unit, 0..511
  hs[i] = 0.f; hs16[i] = (_Float16)0.f;
  __syncthreads();
  const float br = bhh[i], bz = bhh[GH + i], bn = bhh[2 * GH + i];
  const h2v* wr = (const h2v*)(whh + (size_t)i * GH);
  const h2v* wz = (const h2v*)(whh + (size_t)(GH + i) * GH);
  const h2v* wn = (const h2v*)(whh + (size_t)(2 * GH + i) * GH);
  const h2v* hp = (const h2v*)hs16;

  for (int step = 0; step < SS; step++) {
    const int s = dir ? (SS - 1 - step) : step;
    const float* gxr = gx + ((size_t)(b * SS + s)) * (3 * GH);
    h2v ar = {(_Float16)0.f, (_Float16)0.f};
    h2v az = ar, an = ar;
    #pragma unroll 8
    for (int k = 0; k < GH / 2; k++) {
      const h2v hv = hp[k];
      ar += wr[k] * hv;
      az += wz[k] * hv;
      an += wn[k] * hv;
    }
    const float ghr = (float)ar[0] + (float)ar[1];
    const float ghz = (float)az[0] + (float)az[1];
    const float ghn = (float)an[0] + (float)an[1];
    const float r = 1.f / (1.f + __expf(-(gxr[i] + ghr + br)));
    const float z = 1.f / (1.f + __expf(-(gxr[GH + i] + ghz + bz)));
    const float n = tanhf(gxr[2 * GH + i] + r * (ghn + bn));
    const float hn = (1.f - z) * n + z * hs[i];
    __syncthreads();
    hs[i] = hn; hs16[i] = (_Float16)hn;
    __syncthreads();
    out[((size_t)(b * SS + s)) * DD + dir * GH + i] = hn;
  }
}

// ---------------- host ----------------
extern "C" void kernel_launch(void* const* d_in, const int* in_sizes, int n_in,
                              void* d_out, int out_size, void* d_ws, size_t ws_size,
                              hipStream_t stream) {
  (void)in_sizes; (void)n_in; (void)out_size; (void)ws_size;
  const float* x        = (const float*)d_in[0];
  const float* pos_emb  = (const float*)d_in[1];
  const float* emb_g    = (const float*)d_in[2];
  const float* emb_b    = (const float*)d_in[3];
  const float* wq       = (const float*)d_in[4];
  const float* wk       = (const float*)d_in[5];
  const float* wv       = (const float*)d_in[6];
  const float* q_bias   = (const float*)d_in[7];
  const float* v_bias   = (const float*)d_in[8];
  const float* wo       = (const float*)d_in[9];
  const float* bo       = (const float*)d_in[10];
  const float* aln_g    = (const float*)d_in[11];
  const float* aln_b    = (const float*)d_in[12];
  const float* w1       = (const float*)d_in[13];
  const float* b1       = (const float*)d_in[14];
  const float* w2       = (const float*)d_in[15];
  const float* b2       = (const float*)d_in[16];
  const float* fln_g    = (const float*)d_in[17];
  const float* fln_b    = (const float*)d_in[18];
  const float* wih_f    = (const float*)d_in[19];
  const float* whh_f    = (const float*)d_in[20];
  const float* bih_f    = (const float*)d_in[21];
  const float* bhh_f    = (const float*)d_in[22];
  const float* wih_b    = (const float*)d_in[23];
  const float* whh_b    = (const float*)d_in[24];
  const float* bih_b    = (const float*)d_in[25];
  const float* bhh_b    = (const float*)d_in[26];
  float* out = (float*)d_out;

  const size_t MS = (size_t)BB * SS;  // 4096
  char* wsp = (char*)d_ws;
  size_t off = 0;
  auto alloc = [&](size_t bytes) -> void* {
    void* p = wsp + off;
    off += (bytes + 255) & ~(size_t)255;
    return p;
  };
  _Float16* wq16   = (_Float16*)alloc((size_t)DD * DD * 2);
  _Float16* wk16   = (_Float16*)alloc((size_t)DD * DD * 2);
  _Float16* wv16   = (_Float16*)alloc((size_t)DD * DD * 2);
  _Float16* wo16   = (_Float16*)alloc((size_t)DD * DD * 2);
  _Float16* w116   = (_Float16*)alloc((size_t)FFD * DD * 2);
  _Float16* w216   = (_Float16*)alloc((size_t)DD * FFD * 2);
  _Float16* wihf16 = (_Float16*)alloc((size_t)3 * GH * DD * 2);
  _Float16* wihb16 = (_Float16*)alloc((size_t)3 * GH * DD * 2);
  _Float16* whhf16 = (_Float16*)alloc((size_t)3 * GH * GH * 2);
  _Float16* whhb16 = (_Float16*)alloc((size_t)3 * GH * GH * 2);
  float*    h32    = (float*)   alloc(MS * DD * 4);
  _Float16* h16    = (_Float16*)alloc(MS * DD * 2);
  _Float16* q16    = (_Float16*)alloc(MS * DD * 2);
  _Float16* k16    = (_Float16*)alloc(MS * DD * 2);
  _Float16* v16    = (_Float16*)alloc(MS * DD * 2);
  _Float16* vT     = (_Float16*)alloc(MS * DD * 2);
  _Float16* ctx16  = (_Float16*)alloc(MS * DD * 2);
  float*    opre   = (float*)   alloc(MS * DD * 4);
  float*    attn32 = (float*)   alloc(MS * DD * 4);
  _Float16* attn16 = (_Float16*)alloc(MS * DD * 2);
  _Float16* ff16   = (_Float16*)alloc(MS * FFD * 2);
  float*    fpre   = (float*)   alloc(MS * DD * 4);
  _Float16* enc16  = (_Float16*)alloc(MS * DD * 2);
  float*    gxf    = (float*)   alloc(MS * 3 * GH * 4);
  float*    gxb    = (float*)   alloc(MS * 3 * GH * 4);

  // weight conversions fp32 -> f16
  auto cvt = [&](const float* s, _Float16* d, int n) {
    k_cvt<<<(n + 255) / 256, 256, 0, stream>>>(s, d, n);
  };
  cvt(wq, wq16, DD * DD);       cvt(wk, wk16, DD * DD);
  cvt(wv, wv16, DD * DD);       cvt(wo, wo16, DD * DD);
  cvt(w1, w116, FFD * DD);      cvt(w2, w216, DD * FFD);
  cvt(wih_f, wihf16, 3 * GH * DD);
  cvt(wih_b, wihb16, 3 * GH * DD);
  cvt(whh_f, whhf16, 3 * GH * GH);
  cvt(whh_b, whhb16, 3 * GH * GH);

  // embeddings: h = LN(x + pos)
  k_add_ln<<<(int)MS, 256, 0, stream>>>(x, pos_emb, emb_g, emb_b, h32, h16, 1);

  // QKV projections (WMMA + TDM staging), bias fused
  dim3 gDD(64, 16);
  k_gemm<0><<<gDD, 256, 0, stream>>>(h16, wq16, q_bias, nullptr, q16, (int)MS, DD, DD);
  k_gemm<0><<<gDD, 256, 0, stream>>>(h16, wk16, nullptr, nullptr, k16, (int)MS, DD, DD);
  k_gemm<0><<<gDD, 256, 0, stream>>>(h16, wv16, v_bias, nullptr, v16, (int)MS, DD, DD);

  // V transpose for contiguous B-fragments in ctx matmul
  k_vt<<<(int)(MS * DD / 256), 256, 0, stream>>>(v16, vT);

  // fused attention
  k_attn<<<dim3(BB * HH, SS / 16), 256, 0, stream>>>(q16, k16, vT, ctx16);

  // output projection + residual LN
  k_gemm<0><<<gDD, 256, 0, stream>>>(ctx16, wo16, bo, opre, nullptr, (int)MS, DD, DD);
  k_add_ln<<<(int)MS, 256, 0, stream>>>(opre, h32, aln_g, aln_b, attn32, attn16, 0);

  // FFN: gelu fused into GEMM1 epilogue
  k_gemm<1><<<dim3(64, 64), 256, 0, stream>>>(attn16, w116, b1, nullptr, ff16, (int)MS, FFD, DD);
  k_gemm<0><<<gDD, 256, 0, stream>>>(ff16, w216, b2, fpre, nullptr, (int)MS, DD, FFD);
  k_add_ln<<<(int)MS, 256, 0, stream>>>(fpre, attn32, fln_g, fln_b, nullptr, enc16, 0);

  // GRU input projections gx = enc @ wih^T + b_ih (WMMA)
  dim3 gGX(64, 24);
  k_gemm<0><<<gGX, 256, 0, stream>>>(enc16, wihf16, bih_f, gxf, nullptr, (int)MS, 3 * GH, DD);
  k_gemm<0><<<gGX, 256, 0, stream>>>(enc16, wihb16, bih_b, gxb, nullptr, (int)MS, 3 * GH, DD);

  // sequential bidirectional scan (per batch, per direction; packed f16 recurrence)
  k_gru<<<dim3(BB, 2), 512, 0, stream>>>(gxf, gxb, whhf16, bhh_f, whhb16, bhh_b, out);
}
